// GaussianSplatting2D_7928509628611
// MI455X (gfx1250) — matrix operand order repaired and checked by reading, not compiled, not run on hardware
//
#include <hip/hip_runtime.h>
#include <hip/hip_bf16.h>
#include <stdint.h>

typedef __attribute__((ext_vector_type(16))) _Float16 v16h;
typedef __attribute__((ext_vector_type(4)))  _Float16 v4h;
typedef __attribute__((ext_vector_type(8)))  float    v8f;
typedef __attribute__((ext_vector_type(4)))  unsigned int v4u;
typedef __attribute__((ext_vector_type(8)))  int      v8i;
typedef __attribute__((ext_vector_type(4)))  int      v4i;

#define B_      4
#define T_      256
#define D_      768
#define NG      1024        // gaussians per batch
#define NCOLS   36
#define NPAD    48
#define HH      256
#define WW      256

// -0.5 * log2(e): folds exp(-0.5*q) into a single v_exp_f32 (exp2)
#define NEG_HALF_LOG2E (-0.72134752044448169f)

#if defined(__has_builtin)
#if __has_builtin(__builtin_amdgcn_tensor_load_to_lds) && __has_builtin(__builtin_amdgcn_s_wait_tensorcnt)
#define USE_TDM 1
#endif
#endif
#ifndef USE_TDM
#define USE_TDM 0
#endif

__device__ __forceinline__ void pack8(v16h& v, int base, float4 a, float4 b) {
  v[base + 0] = (_Float16)a.x; v[base + 1] = (_Float16)a.y;
  v[base + 2] = (_Float16)a.z; v[base + 3] = (_Float16)a.w;
  v[base + 4] = (_Float16)b.x; v[base + 5] = (_Float16)b.y;
  v[base + 6] = (_Float16)b.z; v[base + 7] = (_Float16)b.w;
}

#if USE_TDM
// 1-D TDM copy: n4 dwords from global gsrc -> LDS offset lds_off.
// D# group0: count=1, lds_addr, 57-bit global addr, type=2 (image).
// D# group1: data_size=4B, tensor_dim0=n4, tensor_dim1=1, tile_dim0=n4,
//            tile_dim1/2 unused, tensor_dim0_stride=n4.
// 6-arg toolchain form: (g0, g1, g2, g3, extra, cpol)
__device__ __forceinline__ void tdm_copy_1d(const void* gsrc,
                                            unsigned int lds_off,
                                            unsigned int n4) {
  unsigned long long ga = (unsigned long long)(uintptr_t)gsrc;
  v4u g0;
  g0.x = 1u;                                        // count=1
  g0.y = lds_off;                                   // lds_addr (bytes)
  g0.z = (unsigned int)ga;                          // global_addr[31:0]
  g0.w = (unsigned int)(ga >> 32) | (2u << 30);     // global_addr[56:32] | type=2
  v8i g1;
  g1[0] = 0x00020000;                               // data_size = 2 (4 bytes)
  g1[1] = (int)((n4 & 0xffffu) << 16);              // tensor_dim0[15:0]
  g1[2] = (int)((n4 >> 16) | (1u << 16));           // tensor_dim0[31:16], tensor_dim1=1
  g1[3] = (int)(n4 << 16);                          // tile_dim0 = n4
  g1[4] = 0;                                        // tile_dim1=0, tile_dim2=0
  g1[5] = (int)n4;                                  // tensor_dim0_stride[31:0]
  g1[6] = 0;
  g1[7] = 0;
  v4i z4 = {0, 0, 0, 0};
  v8i z8 = {0, 0, 0, 0, 0, 0, 0, 0};
  __builtin_amdgcn_tensor_load_to_lds(g0, g1, z4, z4, z8, 0);
}
#endif

// ---------------------------------------------------------------------------
// Kernel 1: projection GEMM  gsraw[row, col] = feature[row,:] . proj_w[col,:]
// M=1024 rows, N=36 cols (padded to 48), K=768. One 16x16 tile per wave.
// Columns >=36 compute garbage into the padded region (never read) -> no
// divergence, no zero-fill in the K loop.
// ---------------------------------------------------------------------------
__global__ __launch_bounds__(256)
void k_proj(const float* __restrict__ feature,
            const float* __restrict__ proj_w,
            float* __restrict__ gsraw) {
  const int lane  = threadIdx.x & 31;
  const int wid   = blockIdx.x * 8 + (threadIdx.x >> 5);  // 0..191
  const int tileN = wid >> 6;                             // 0..2
  const int tileM = wid & 63;                             // 0..63
  const int l16   = lane & 15;
  const int half  = lane >> 4;                            // 0 or 1

  const int row    = tileM * 16 + l16;
  const int ncol   = tileN * 16 + l16;
  const int ncol_c = (ncol < NCOLS) ? ncol : (NCOLS - 1);  // clamp, no branch

  v8f acc = {};
  for (int k0 = 0; k0 < D_; k0 += 32) {
    // A fragment: lane<16 holds K = k0+{0..7,16..23}; lane>=16 shifted by +8
    const int ka = k0 + half * 8;
    const float4* fa = (const float4*)(feature + row * D_ + ka);
    float4 a0 = fa[0], a1 = fa[1];
    const float4* fb = (const float4*)(feature + row * D_ + ka + 16);
    float4 a2 = fb[0], a3 = fb[1];
    v16h A;
    pack8(A, 0, a0, a1);
    pack8(A, 8, a2, a3);

    // B fragment: B[k][n] = proj_w[n][k]; lane<16 K = k0+{0..7,16..23}; lane>=16 shifted by +8
    const int kb = k0 + half * 8;
    const float4* pb = (const float4*)(proj_w + ncol_c * D_ + kb);
    float4 b0 = pb[0], b1 = pb[1], b2 = pb[4], b3 = pb[5];
    v16h Bm;
    pack8(Bm, 0, b0, b1);
    pack8(Bm, 8, b2, b3);

    acc = __builtin_amdgcn_wmma_f32_16x16x32_f16(false, A, false, Bm,
                                                 (short)0, acc, false, false);
  }

  // C layout: VGPR r of lane holds C[M = r + 8*half][N = l16]
  const int rowbase = tileM * 16 + half * 8;
#pragma unroll
  for (int r = 0; r < 8; ++r)
    gsraw[(rowbase + r) * NPAD + ncol] = acc[r];
}

// ---------------------------------------------------------------------------
// Kernel 2: per-gaussian parameter transform (4096 gaussians)
//   gw[g]   = {px, py, c*i00, c*2*i01, c*i11, log2(alpha), 0, 0}  (fp32 x8)
//             with c = -0.5*log2(e)  -> splat needs a single exp2 per pair
//   colh[g] = {r, g, b, 1.0}                                      (f16 x4)
// ---------------------------------------------------------------------------
__global__ __launch_bounds__(256)
void k_params(const float* __restrict__ gsraw,
              const float* __restrict__ proj_b,
              float* __restrict__ gw,
              v4h* __restrict__ colh) {
  const int tid = blockIdx.x * blockDim.x + threadIdx.x;   // 0..4095
  const int b = tid >> 10, n = tid & 1023;
  const int t = n >> 2,  j = n & 3;
  const int row = b * T_ + t;

  const float* g  = gsraw + row * NPAD + j * 9;
  const float* bb = proj_b + j * 9;
  float v[9];
#pragma unroll
  for (int p = 0; p < 9; ++p) v[p] = g[p] + bb[p];

  float s0 = __expf(v[2]), s1 = __expf(v[3]);
  float sx2 = s0 * s0, sy2 = s1 * s1;
  float sn, cs;
  __sincosf(v[4], &sn, &cs);
  // log2(sigmoid(x)) = -log2(1 + exp(-x))
  float la  = -__log2f(1.0f + __expf(-v[5]));
  float c00 = cs * cs * sx2 + sn * sn * sy2 + 1e-6f;
  float c01 = cs * sn * (sx2 - sy2);
  float c11 = sn * sn * sx2 + cs * cs * sy2 + 1e-6f;
  float invdet = 1.0f / (c00 * c11 - c01 * c01);
  float i00 =  c11 * invdet;
  float i01 = -c01 * invdet;
  float i11 =  c00 * invdet;

  float4* o = (float4*)(gw + tid * 8);
  o[0] = make_float4(v[0], v[1], NEG_HALF_LOG2E * i00, NEG_HALF_LOG2E * 2.0f * i01);
  o[1] = make_float4(NEG_HALF_LOG2E * i11, la, 0.0f, 0.0f);

  v4h c;
  c[0] = (_Float16)v[6]; c[1] = (_Float16)v[7]; c[2] = (_Float16)v[8];
  c[3] = (_Float16)1.0f;
  colh[tid] = c;
}

// ---------------------------------------------------------------------------
// Kernel 3: splat. One block per (batch, image row). 512 threads = 16 waves,
// each wave owns 16 pixels. Params broadcast via LDS (filled by TDM when
// available); per chunk of 32 gaussians the wave computes a 16x32 f16 weight
// fragment (7 VALU + 1 v_exp_f32 per pair) and one v_wmma_f32_16x16x32_f16
// accumulates colors + weight-sum.
// LDS map (dynamic): [0, 32768)  = gaussian params (1024 x 8 f32)
//                    [32768, +8K)= colors          (1024 x 4 f16)
// ---------------------------------------------------------------------------
__global__ __launch_bounds__(512)
void k_splat(const float* __restrict__ gw,
             const v4h* __restrict__ colh,
             float* __restrict__ out) {
  extern __shared__ __align__(16) char smem[];
  float*    sgw  = (float*)smem;                 // 32 KB
  _Float16* scol = (_Float16*)(smem + 32768);    //  8 KB

  const int b   = blockIdx.x >> 8;
  const int h   = blockIdx.x & 255;
  const int tid = threadIdx.x;

#if USE_TDM
  if (tid < 32) {   // wave 0 issues both DMA descriptors (EXEC-independent)
    tdm_copy_1d(gw + (size_t)b * NG * 8, 0u,     NG * 2);  // 2048 dwords
    tdm_copy_1d(colh + (size_t)b * NG,   32768u, 2048);    // 2048 dwords
    __builtin_amdgcn_s_wait_tensorcnt(0);
  }
  __syncthreads();
#else
  {
    const float4* src = (const float4*)(gw + (size_t)b * NG * 8);
    float4* dst = (float4*)sgw;
    for (int i = tid; i < NG * 2; i += 512) dst[i] = src[i];
    const float4* cs4 = (const float4*)(colh + (size_t)b * NG);
    float4* cd = (float4*)scol;
    for (int i = tid; i < NG / 2; i += 512) cd[i] = cs4[i];
  }
  __syncthreads();
#endif

  const int lane = tid & 31;
  const int wv   = tid >> 5;        // 0..15
  const int l16  = lane & 15;
  const int half = lane >> 4;
  const int w0   = wv * 16;

  const float x = -1.0f + (2.0f / 255.0f) * (float)(w0 + l16);
  const float y = -1.0f + (2.0f / 255.0f) * (float)h;

  v8f acc = {};
  for (int chunk = 0; chunk < NG; chunk += 32) {
    // ---- A fragment: weights for pixel l16, 16 gaussians of this half-wave
    v16h A;
    const int ga = chunk + half * 8;
#pragma unroll
    for (int e = 0; e < 16; ++e) {
      const int g = ga + ((e < 8) ? e : (e + 8));
      const float4* P = (const float4*)(sgw + g * 8);
      float4 p0 = P[0];              // px, py, i00s, i01s2
      float4 p1 = P[1];              // i11s, log2(alpha), 0, 0
      float dx = x - p0.x;
      float dy = y - p0.y;
      float t = __builtin_fmaf(p0.w, dy, p0.z * dx);        // i00s*dx + i01s2*dy
      float u = __builtin_fmaf(p1.x * dy, dy, p1.y);        // i11s*dy^2 + log2a
      float q = __builtin_fmaf(dx, t, u);
      A[e] = (_Float16)__builtin_amdgcn_exp2f(q);
    }
    // ---- B fragment: column n = l16 of {r,g,b,1}; K in same order as A
    v16h Bm;
    const int gb = chunk + half * 8;
#pragma unroll
    for (int e = 0; e < 16; ++e) {
      _Float16 val = scol[(gb + ((e < 8) ? e : (e + 8))) * 4 + (l16 & 3)];
      Bm[e] = (l16 < 4) ? val : (_Float16)0.0f;
    }
    acc = __builtin_amdgcn_wmma_f32_16x16x32_f16(false, A, false, Bm,
                                                 (short)0, acc, false, false);
  }

  // Epilogue: column 3 of C holds sum of weights; broadcast from lanes 3 / 19.
  const int srcLane = (lane & 16) | 3;
  float tmp[8];
#pragma unroll
  for (int r = 0; r < 8; ++r) {
    float wsr = __shfl(acc[r], srcLane, 32);
    tmp[r] = acc[r] / (wsr + 1e-6f);
  }
  if (l16 < 3) {
    float* o = out + ((((size_t)b * 3 + l16) * HH + h) * WW + w0 + half * 8);
    float4* o4 = (float4*)o;
    o4[0] = make_float4(tmp[0], tmp[1], tmp[2], tmp[3]);
    o4[1] = make_float4(tmp[4], tmp[5], tmp[6], tmp[7]);
  }
}

// ---------------------------------------------------------------------------
extern "C" void kernel_launch(void* const* d_in, const int* in_sizes, int n_in,
                              void* d_out, int out_size, void* d_ws, size_t ws_size,
                              hipStream_t stream) {
  const float* feature = (const float*)d_in[0];   // (4,256,768) f32
  const float* proj_w  = (const float*)d_in[1];   // (36,768)    f32
  const float* proj_b  = (const float*)d_in[2];   // (36,)       f32
  float* out = (float*)d_out;                     // (4,3,256,256) f32

  char* ws = (char*)d_ws;
  float* gsraw = (float*)ws;                                   // 1024*48 f32 = 192 KB
  float* gwbuf = (float*)(ws + (size_t)1024 * NPAD * 4);       // 4096*8  f32 = 128 KB
  v4h*   cbuf  = (v4h*)  (ws + (size_t)1024 * NPAD * 4
                              + (size_t)4096 * 8 * 4);         // 4096*4  f16 =  32 KB

  k_proj  <<<24,   256, 0, stream>>>(feature, proj_w, gsraw);
  k_params<<<16,   256, 0, stream>>>(gsraw, proj_b, gwbuf, cbuf);
  k_splat <<<1024, 512, 40960, stream>>>(gwbuf, cbuf, out);
}
